// Spectrogram_67207648248372
// MI455X (gfx1250) — compile-verified, matrix operand order checked by reading, not run on hardware
//
#include <hip/hip_runtime.h>
#include <hip/hip_bf16.h>

// ---------------- problem constants ----------------
#define NB      32          // batch
#define NC      2           // channels
#define NL      262144      // samples per channel
#define NDFT    512
#define NF      257         // n_dft/2 + 1
#define NFPAD   272         // 17 * 16
#define NHOP    256
#define NT      1024        // frames per channel
#define LPAD    262400      // 128 + NL + 128  (pad_l = pad_r = 128)
#define KSTEPS  16          // 512 / 32
#define AMIN    1e-10f
#define DRANGE  80.0f

// LDS B-tile geometry: [mat 2][col 16][k 512] with col stride padded to 520 halves
#define LBSTRIDE 520
#define LBMAT    (16 * LBSTRIDE)   // halves per one matrix tile (16 cols)
#define LBBUF    (2 * LBMAT)       // halves per buffer (re + im)

typedef _Float16 v16h __attribute__((ext_vector_type(16)));
typedef _Float16 v8h  __attribute__((ext_vector_type(8)));
typedef float    v8f  __attribute__((ext_vector_type(8)));
typedef unsigned v4u  __attribute__((ext_vector_type(4)));
typedef int      v4i  __attribute__((ext_vector_type(4)));
typedef int      v8i  __attribute__((ext_vector_type(8)));

union V16 { v16h v; v8h h[2]; };

#if defined(__gfx1250__) && __has_builtin(__builtin_amdgcn_tensor_load_to_lds) && \
    __has_builtin(__builtin_amdgcn_s_wait_tensorcnt)
#define HAVE_TDM 1
#else
#define HAVE_TDM 0
#endif

// ---------------- kernel 1: pack DFT kernels (f-major, f16, zero-padded cols) ----------------
// Bt[m][f][k] = (f < 257) ? K_m[k*257 + f] : 0     shape [2][272][512] f16
__global__ __launch_bounds__(256) void pack_k_kernel(const float* __restrict__ kr,
                                                     const float* __restrict__ ki,
                                                     _Float16* __restrict__ Bt,
                                                     unsigned* __restrict__ maxbits) {
    int i = blockIdx.x * 256 + threadIdx.x;
    if (i == 0) *maxbits = 0u;                       // init global-max (power >= 0)
    if (i >= 2 * NFPAD * NDFT) return;
    int m = i / (NFPAD * NDFT);
    int r = i % (NFPAD * NDFT);
    int f = r / NDFT;
    int k = r % NDFT;
    float v = 0.0f;
    if (f < NF) v = (m ? ki : kr)[k * NF + f];
    Bt[i] = (_Float16)v;
}

// ---------------- kernel 2: pack input to padded f16 ----------------
__global__ __launch_bounds__(256) void pack_x_kernel(const float* __restrict__ x,
                                                     _Float16* __restrict__ xh) {
    int i = blockIdx.x * 256 + threadIdx.x;
    if (i >= NB * NC * LPAD) return;
    int bc = i / LPAD;
    int p  = i % LPAD;
    int s  = p - 128;
    float v = 0.0f;
    if (s >= 0 && s < NL) v = x[(size_t)bc * NL + s];
    xh[i] = (_Float16)v;
}

#if HAVE_TDM
// Numeric LDS byte offset of a generic pointer into shared memory.
__device__ __forceinline__ unsigned lds_off_of(const void* p) {
    return (unsigned)(unsigned long long)(__attribute__((address_space(3))) const void*)p;
}

// Issue TDM loads of both DFT-kernel tiles (re + im) for one f-tile into LDS.
// Tile: 16 rows (f) x 512 cols (k) of 2-byte data, row stride 512 elems in memory,
// LDS padding: 4 DWORDs after every 256 DWORDs -> row stride 520 halves in LDS.
__device__ __forceinline__ void tdm_issue_btile(const _Float16* __restrict__ Bt,
                                                int fbase, unsigned ldsByteOff) {
#pragma unroll
    for (int m = 0; m < 2; ++m) {
        unsigned long long ga =
            (unsigned long long)(size_t)(Bt + ((size_t)(m * NFPAD + fbase) * NDFT));
        v4u g0;
        g0.x = 1u;                                            // count=1, user desc
        g0.y = ldsByteOff + (unsigned)(m * LBMAT * 2);        // lds_addr (bytes)
        g0.z = (unsigned)(ga & 0xFFFFFFFFull);                // global_addr[31:0]
        g0.w = (unsigned)((ga >> 32) & 0x01FFFFFFull)         // global_addr[56:32]
               | 0x80000000u;                                 // type=2 ("image")
        v8i g1;
        g1[0] = (1 << 16)      // data_size = 1 -> 2 bytes
              | (1 << 20)      // pad_enable
              | (7 << 22)      // pad_interval = 7 -> every 256 DWORDs (one 512-half row)
              | (3 << 25);     // pad_amount   = 3 -> 4 DWORDs (row stride 520 halves)
        g1[1] = (int)(512u << 16);   // tensor_dim0[15:0] = 512 (bits 79:48)
        g1[2] = (int)(16u << 16);    // tensor_dim0 hi = 0 | tensor_dim1[15:0] = 16
        g1[3] = (int)(512u << 16);   // tensor_dim1 hi = 0 | tile_dim0 = 512
        g1[4] = 16;                  // tile_dim1 = 16, tile_dim2 = 0 (unused)
        g1[5] = 512;                 // tensor_dim0_stride[31:0] = 512 elems
        g1[6] = (int)(512u << 16);   // stride0 hi = 0 | tensor_dim1_stride[15:0] = 512
        g1[7] = 0;
        v4i gz4 = {0, 0, 0, 0};                    // 2-D tensor: groups 2/3 unused
        v8i gz8 = {0, 0, 0, 0, 0, 0, 0, 0};        // unused trailing group (clang-23 form)
        __builtin_amdgcn_tensor_load_to_lds(g0, g1, gz4, gz4, gz8, 0);
    }
}
#endif

// ---------------- kernel 3: WMMA GEMM + power + transposed store + global max ----------------
// grid.x = 64 (b,c) * 8 slabs = 512 blocks; 256 threads = 8 waves; wave w owns a 16-frame M-tile.
__global__ __launch_bounds__(256) void spect_gemm_kernel(const _Float16* __restrict__ xh,
                                                         const _Float16* __restrict__ Bt,
                                                         float* __restrict__ out,
                                                         unsigned* __restrict__ maxbits) {
#if HAVE_TDM
    __shared__ __align__(16) _Float16 lB[2 * LBBUF];   // ping-pong: 66,560 B of 320 KB LDS
#else
    __shared__ __align__(16) _Float16 lB[LBBUF];
#endif

    const int bc   = blockIdx.x >> 3;        // 0..63
    const int slab = blockIdx.x & 7;         // 0..7  (128 frames each)
    const int b    = bc >> 1;
    const int c    = bc & 1;
    const int wave = threadIdx.x >> 5;
    const int lane = threadIdx.x & 31;
    const int lo   = lane & 15;
    const int hi   = lane >> 4;
    const int t0   = slab * 128 + wave * 16;

    // A fragment base: lane holds row M = lo, K-chunk offset 8*hi (ISA 16-bit A layout)
    const _Float16* aBase = xh + (size_t)bc * LPAD + (size_t)(t0 + lo) * NHOP + 8 * hi;

    float mloc = 0.0f;

#if HAVE_TDM
    // Prefetch f-tile 0 into buffer 0 (DMA; no VGPR traffic, tracked on TENSORcnt)
    if (wave == 0) tdm_issue_btile(Bt, 0, lds_off_of(lB));
#endif

    for (int ft = 0; ft < 17; ++ft) {
        const int fbase = ft * 16;
#if HAVE_TDM
        __builtin_amdgcn_s_wait_tensorcnt(0);    // wave0: current buffer DMA complete
        __syncthreads();                          // everyone: done reading prev buffer
        const _Float16* buf = lB + (size_t)(ft & 1) * LBBUF;
        if (wave == 0 && ft + 1 < 17)             // overlap: DMA next tile while computing
            tdm_issue_btile(Bt, (ft + 1) * 16,
                            lds_off_of(lB) + (unsigned)(((ft + 1) & 1) * LBBUF * 2));
#else
        __syncthreads();
        // --- stage B_re and B_im tiles (2 x 512 x 16 halves) into LDS manually ---
        #pragma unroll
        for (int it = 0; it < 8; ++it) {
            int chunk = it * 256 + threadIdx.x;      // 2048 chunks of 8 halves
            int m  = chunk >> 10;
            int r  = chunk & 1023;
            int n  = r >> 6;
            int k8 = (r & 63) * 8;
            *(v8h*)&lB[(m * 16 + n) * LBSTRIDE + k8] =
                *(const v8h*)&Bt[((size_t)(m * NFPAD + fbase + n) * NDFT) + k8];
        }
        __syncthreads();
        const _Float16* buf = lB;
#endif

        v8f accR = {};
        v8f accI = {};
        // B fragment base: lane holds col N = lo, K-chunk offset 8*hi (mirror of A layout)
        const _Float16* bR = &buf[(0 * 16 + lo) * LBSTRIDE + 8 * hi];
        const _Float16* bI = &buf[(1 * 16 + lo) * LBSTRIDE + 8 * hi];

        #pragma unroll
        for (int s = 0; s < KSTEPS; ++s) {
            V16 a, br, bi;
            a.h[0]  = *(const v8h*)(aBase + s * 32);
            a.h[1]  = *(const v8h*)(aBase + s * 32 + 16);
            br.h[0] = *(const v8h*)(bR + s * 32);
            br.h[1] = *(const v8h*)(bR + s * 32 + 16);
            bi.h[0] = *(const v8h*)(bI + s * 32);
            bi.h[1] = *(const v8h*)(bI + s * 32 + 16);
            accR = __builtin_amdgcn_wmma_f32_16x16x32_f16(false, a.v, false, br.v,
                                                          (short)0, accR, false, false);
            accI = __builtin_amdgcn_wmma_f32_16x16x32_f16(false, a.v, false, bi.v,
                                                          (short)0, accI, false, false);
        }

        // --- power, transposed store out[b][f][t][c], track max ---
        const int f = fbase + lo;
        if (f < NF) {
            // C/D layout: VGPR j -> M = j + 8*hi, N = lo
            size_t obase = (((size_t)b * NF + f) * NT + (t0 + hi * 8)) * NC + c;
            #pragma unroll
            for (int j = 0; j < 8; ++j) {
                float p = accR[j] * accR[j] + accI[j] * accI[j];
                mloc = fmaxf(mloc, p);
                out[obase + (size_t)j * NC] = p;
            }
        }
    }

    // wave32 max reduction; power >= 0 so IEEE order == uint order
    #pragma unroll
    for (int off = 16; off > 0; off >>= 1)
        mloc = fmaxf(mloc, __shfl_xor(mloc, off, 32));
    if (lane == 0) atomicMax(maxbits, __float_as_uint(mloc));
}

// ---------------- kernel 4: in-place dB conversion, normalize, clamp ----------------
__global__ __launch_bounds__(256) void finalize_kernel(float* __restrict__ out,
                                                       const unsigned* __restrict__ maxbits,
                                                       int total) {
    int i = blockIdx.x * 256 + threadIdx.x;
    if (i >= total) return;
    float mp  = fmaxf(__uint_as_float(*maxbits), AMIN);
    float ref = 10.0f * log10f(mp);
    float p   = fmaxf(out[i], AMIN);
    out[i] = fmaxf(10.0f * log10f(p) - ref, -DRANGE);
}

// ---------------- host launcher ----------------
extern "C" void kernel_launch(void* const* d_in, const int* in_sizes, int n_in,
                              void* d_out, int out_size, void* d_ws, size_t ws_size,
                              hipStream_t stream) {
    const float* x  = (const float*)d_in[0];
    const float* kr = (const float*)d_in[1];
    const float* ki = (const float*)d_in[2];
    float* out = (float*)d_out;

    char* ws = (char*)d_ws;
    unsigned*  maxbits = (unsigned*)ws;                                  // 4 B (256 reserved)
    _Float16*  Bt      = (_Float16*)(ws + 256);                          // 2*272*512*2 = 557056 B
    _Float16*  xh      = (_Float16*)(ws + 256 + 2 * NFPAD * NDFT * 2);   // 64*262400*2 B

    // 1) transpose + f16-pack DFT kernels, init max
    {
        int total = 2 * NFPAD * NDFT;
        pack_k_kernel<<<(total + 255) / 256, 256, 0, stream>>>(kr, ki, Bt, maxbits);
    }
    // 2) zero-pad + f16-pack input
    {
        int total = NB * NC * LPAD;
        pack_x_kernel<<<(total + 255) / 256, 256, 0, stream>>>(x, xh);
    }
    // 3) WMMA spectrogram GEMM (power into d_out, transposed layout) + global max
    spect_gemm_kernel<<<NB * NC * 8, 256, 0, stream>>>(xh, Bt, out, maxbits);
    // 4) log-power dB, normalize by global max, clamp
    {
        int total = NB * NF * NT * NC;   // 16,842,752
        finalize_kernel<<<(total + 255) / 256, 256, 0, stream>>>(out, maxbits, total);
    }
}
